// LMulApproxLinearBF16_43602507989103
// MI455X (gfx1250) — matrix operand order repaired and checked by reading, not compiled, not run on hardware
//
#include <hip/hip_runtime.h>
#include <hip/hip_bf16.h>

// L-Mul approximate linear layer on MI455X (gfx1250), mapped onto bf16 WMMA.
//
// Identity: for bf16 x = X0 + X1 with X0 = sign*2^e (bits & 0xFF80), X1 = s*m'*2^e,
//   lmul(x,w) = (1+eps)*X0*W0 + X1*W0 + X0*W1 = X0*((1+eps)W0 + W1) + X1*W0
// => out = X0 @ Wc^T + X1 @ W0^T + bias, two dense bf16 GEMMs, one f32 accumulator.

typedef __attribute__((ext_vector_type(16))) __bf16        v16bf;
typedef __attribute__((ext_vector_type(8)))  float         v8f;
typedef __attribute__((ext_vector_type(4)))  unsigned int  u32x4;

union FragCvt {
  v16bf v;
  u32x4 q[2];
};

__device__ __forceinline__ float bfbits_to_f32(unsigned int b) {
  unsigned int u = b << 16;
  float f;
  __builtin_memcpy(&f, &u, sizeof(f));
  return f;
}

__device__ __forceinline__ unsigned short f32_to_bfbits(float f) {
  unsigned int u;
  __builtin_memcpy(&u, &f, sizeof(u));
  unsigned int r = u + 0x7FFFu + ((u >> 16) & 1u);  // round-to-nearest-even
  return (unsigned short)(r >> 16);
}

// ---------------------------------------------------------------------------
// Prep kernels: split bf16 into exponent part (X0 = bits & 0xFF80) and
// mantissa part (X1 = x - X0, exact in bf16). For W additionally fold the
// L-Mul epsilon term: Wc = (1+2^-4)*W0 + W1.
// ---------------------------------------------------------------------------

__global__ __launch_bounds__(256) void lmul_prep_x(
    const unsigned short* __restrict__ x,
    unsigned short* __restrict__ X0,
    unsigned short* __restrict__ X1,
    int n) {
  int i = blockIdx.x * blockDim.x + threadIdx.x;
  if (i >= n) return;
  unsigned short b  = x[i];
  unsigned short b0 = b & 0xFF80u;
  X0[i] = b0;
  float xf  = bfbits_to_f32(b);
  float x0f = bfbits_to_f32(b0);
  X1[i] = f32_to_bfbits(xf - x0f);  // exact: <=7 significant mantissa bits
}

__global__ __launch_bounds__(256) void lmul_prep_w(
    const unsigned short* __restrict__ w,
    unsigned short* __restrict__ W0,
    unsigned short* __restrict__ Wc,
    int n) {
  int i = blockIdx.x * blockDim.x + threadIdx.x;
  if (i >= n) return;
  unsigned short b  = w[i];
  unsigned short b0 = b & 0xFF80u;
  W0[i] = b0;
  float wf  = bfbits_to_f32(b);
  float w0f = bfbits_to_f32(b0);
  // (1 + 2^-4) * W0 + W1  (L-Mul epsilon folded into the B operand)
  Wc[i] = f32_to_bfbits(1.0625f * w0f + (wf - w0f));
}

// ---------------------------------------------------------------------------
// Fragment loaders matching the documented gfx1250 WMMA VGPR layouts.
// A (16x32 bf16, M x K): lanes 0-15 hold row M=lane, K in {0..7, 16..23};
//                        lanes 16-31 hold row M=lane-16, K in {8..15, 24..31}.
// B (32x16 bf16, K x N): lanes 0-15 hold col N=lane, K in {0..7, 16..23};
//                        lanes 16-31 hold col N=lane-16, K in {8..15, 24..31}.
// Both X[N,K] and W[O,K] are K-major, so every per-lane chunk is contiguous.
// ---------------------------------------------------------------------------

__device__ __forceinline__ v16bf load_a_frag(const unsigned short* __restrict__ X,
                                             int ld, int row0, int k0, int lane) {
  const int half = lane >> 4;
  const int r    = lane & 15;
  const unsigned short* p = X + (size_t)(row0 + r) * ld + k0 + half * 8;
  FragCvt f;
  f.q[0] = *reinterpret_cast<const u32x4*>(p);        // K 0..7  (or 8..15)
  f.q[1] = *reinterpret_cast<const u32x4*>(p + 16);   // K 16..23 (or 24..31)
  return f.v;
}

__device__ __forceinline__ v16bf load_b_frag(const unsigned short* __restrict__ W,
                                             int ld, int col0, int k0, int lane) {
  const int half = lane >> 4;
  const int n    = lane & 15;
  const unsigned short* p = W + (size_t)(col0 + n) * ld + k0 + half * 8;
  FragCvt f;
  f.q[0] = *reinterpret_cast<const u32x4*>(p);        // K 0..7  (or 8..15)
  f.q[1] = *reinterpret_cast<const u32x4*>(p + 16);   // K 16..23 (or 24..31)
  return f.v;
}

// ---------------------------------------------------------------------------
// GEMM: out[m,o] = sum_k X0[m,k]*Wc[o,k] + X1[m,k]*W0[o,k] + bias[o]
// Block = 256 threads = 8 wave32s arranged 2(M) x 4(O).
// Wave tile = 32x64 -> 2x4 accumulator tiles of 16x16 f32.
// Block tile = 64 x 256. 16 v_wmma per K-step per wave.
// ---------------------------------------------------------------------------

#define TBM 64
#define TBO 256
#define TWM 32
#define TWO 64

__global__ __launch_bounds__(256) void lmul_wmma_gemm(
    const unsigned short* __restrict__ X0,
    const unsigned short* __restrict__ X1,
    const unsigned short* __restrict__ Wc,
    const unsigned short* __restrict__ W0,
    const unsigned short* __restrict__ bias,
    unsigned short* __restrict__ out,
    int N, int O, int K) {
  const int lane = threadIdx.x & 31;
  const int wv   = threadIdx.x >> 5;
  const int m0   = blockIdx.x * TBM + (wv >> 2) * TWM;  // wave row base
  const int o0   = blockIdx.y * TBO + (wv & 3) * TWO;   // wave col base

  v8f acc[2][4];
#pragma unroll
  for (int mi = 0; mi < 2; ++mi)
#pragma unroll
    for (int oi = 0; oi < 4; ++oi)
      acc[mi][oi] = (v8f){0.f, 0.f, 0.f, 0.f, 0.f, 0.f, 0.f, 0.f};

  for (int k0 = 0; k0 < K; k0 += 32) {
    // Prefetch the next K slab (emits global_prefetch_b8).
    if (k0 + 32 < K) {
      __builtin_prefetch(X0 + (size_t)(m0 + (lane & 15)) * K + k0 + 32, 0, 0);
      __builtin_prefetch(Wc + (size_t)(o0 + lane) * K + k0 + 32, 0, 0);
    }

    v16bf a0[2], a1[2], bc[4], b0[4];
#pragma unroll
    for (int mi = 0; mi < 2; ++mi) {
      a0[mi] = load_a_frag(X0, K, m0 + 16 * mi, k0, lane);
      a1[mi] = load_a_frag(X1, K, m0 + 16 * mi, k0, lane);
    }
#pragma unroll
    for (int oi = 0; oi < 4; ++oi) {
      bc[oi] = load_b_frag(Wc, K, o0 + 16 * oi, k0, lane);
      b0[oi] = load_b_frag(W0, K, o0 + 16 * oi, k0, lane);
    }

#pragma unroll
    for (int mi = 0; mi < 2; ++mi) {
#pragma unroll
      for (int oi = 0; oi < 4; ++oi) {
        acc[mi][oi] = __builtin_amdgcn_wmma_f32_16x16x32_bf16(
            false, a0[mi], false, bc[oi], (short)0, acc[mi][oi], false, false);
        acc[mi][oi] = __builtin_amdgcn_wmma_f32_16x16x32_bf16(
            false, a1[mi], false, b0[oi], (short)0, acc[mi][oi], false, false);
      }
    }
  }

  // Epilogue: C/D layout -> VGPR r holds M = r + (lane>=16 ? 8 : 0), N = lane&15.
  const int half = lane >> 4;
  const int n    = lane & 15;
#pragma unroll
  for (int oi = 0; oi < 4; ++oi) {
    const int oc = o0 + 16 * oi + n;
    const float bv = bfbits_to_f32(bias[oc]);
#pragma unroll
    for (int mi = 0; mi < 2; ++mi) {
#pragma unroll
      for (int r = 0; r < 8; ++r) {
        const int m = m0 + 16 * mi + r + half * 8;
        out[(size_t)m * O + oc] = f32_to_bfbits(acc[mi][oi][r] + bv);
      }
    }
  }
}

// ---------------------------------------------------------------------------

extern "C" void kernel_launch(void* const* d_in, const int* in_sizes, int n_in,
                              void* d_out, int out_size, void* d_ws, size_t ws_size,
                              hipStream_t stream) {
  const unsigned short* x    = (const unsigned short*)d_in[0];  // bf16 [N,K]
  const unsigned short* wgt  = (const unsigned short*)d_in[1];  // bf16 [O,K]
  const unsigned short* bias = (const unsigned short*)d_in[2];  // bf16 [1,O]
  unsigned short*       out  = (unsigned short*)d_out;          // bf16 [N,O]

  const int O = in_sizes[2];            // 1024
  const int K = in_sizes[1] / O;        // 1024
  const int N = in_sizes[0] / K;        // 512

  // Workspace layout (6 MiB total for the reference sizes).
  unsigned short* X0 = (unsigned short*)d_ws;
  unsigned short* X1 = X0 + (size_t)N * K;
  unsigned short* W0 = X1 + (size_t)N * K;
  unsigned short* Wc = W0 + (size_t)O * K;

  const int nx = N * K;
  const int nw = O * K;
  lmul_prep_x<<<(nx + 255) / 256, 256, 0, stream>>>(x, X0, X1, nx);
  lmul_prep_w<<<(nw + 255) / 256, 256, 0, stream>>>(wgt, W0, Wc, nw);

  dim3 grid((N + TBM - 1) / TBM, (O + TBO - 1) / TBO);  // (8, 4)
  lmul_wmma_gemm<<<grid, 256, 0, stream>>>(X0, X1, Wc, W0, bias, out, N, O, K);
}
